// SelfConnectionIntro_86251533238569
// MI455X (gfx1250) — matrix operand order repaired and checked by reading, not compiled, not run on hardware
//
#include <hip/hip_runtime.h>

// MI455X / gfx1250, wave32. f16 WMMA (16x16x32, f32 accum) GEMM formulation of
// the e3nn FullyConnectedTensorProduct with all-scalar z operand:
//   out[n, w*d+m] = alpha * sum_{u,v} x[n,u,m] * z[n,v] * w[u,v,w]
// = GEMM with A[n, (u,v)] = x[n,u,m]*z[n,v] built on the fly, B = w (stationary).
// V padded 89->96 so each K-chunk of 32 has a single u: A_frag = x_scalar * z_frag.
// l=0 block uses 2 node-subtiles per wave (256 nodes/WG) to halve L2 weight
// traffic and run 2 WMMAs per B-fragment load.

typedef __attribute__((ext_vector_type(16))) _Float16 v16h;
typedef __attribute__((ext_vector_type(8)))  float    v8f;

#define NSP   89      // species (K inner dim before padding)
#define KPAD  96      // padded to 3 chunks of 32
#define XDIM  480     // 128*1 + 64*3 + 32*5

// Fragment counts per l-block: U * 3 chunks * (W/16) tiles
#define FRAGS0 (128 * 3 * 8)   // 3072
#define FRAGS1 (64  * 3 * 4)   // 768
#define FRAGS2 (32  * 3 * 2)   // 192
#define FRAGS_TOTAL (FRAGS0 + FRAGS1 + FRAGS2)   // 4032, 1KB each

static __device__ inline v16h splat16(_Float16 s) {
  v16h v;
#pragma unroll
  for (int i = 0; i < 16; ++i) v[i] = s;
  return v;
}

// ---------------------------------------------------------------------------
// pack_w: f32 path weights -> f16 B-fragments.
// B tile = 32(K) x 16(N). Lane L: column = L%16, K half = L/16.
// Per lane 16 contiguous f16, j in [0,16): K = (L/16)*16 + j.
// Frag id within block: (u*3 + c)*T + t,  T = W/16, c = 32-wide K chunk.
// ---------------------------------------------------------------------------
__global__ __launch_bounds__(256) void pack_w_kernel(
    const float* __restrict__ w0, const float* __restrict__ w1,
    const float* __restrict__ w2, _Float16* __restrict__ bw, int totalElems) {
  int e = blockIdx.x * blockDim.x + threadIdx.x;
  if (e >= totalElems) return;
  int j    = e & 15;
  int lane = (e >> 4) & 31;
  int fragG = e >> 9;              // global fragment id

  const float* w;
  int W, f;
  if (fragG < FRAGS0)                { w = w0; W = 128; f = fragG; }
  else if (fragG < FRAGS0 + FRAGS1)  { w = w1; W = 64;  f = fragG - FRAGS0; }
  else                               { w = w2; W = 32;  f = fragG - FRAGS0 - FRAGS1; }
  int T   = W >> 4;
  int u   = f / (3 * T);
  int rem = f % (3 * T);
  int c   = rem / T;
  int t   = rem % T;

  int h    = lane >> 4;
  int k    = c * 32 + h * 16 + j;        // padded species index
  int wcol = t * 16 + (lane & 15);
  float val = (k < NSP) ? w[((size_t)u * NSP + k) * W + wcol] : 0.0f;
  bw[(size_t)fragG * 512 + lane * 16 + j] = (_Float16)val;
}

// ---------------------------------------------------------------------------
// Main fused kernel, one instantiation per l-block.
// S = node-subtiles per wave (16 nodes each), NPW = S*128 nodes per workgroup.
// U = mul_x = mul_out, W = mul_out, D = 2l+1.
// ---------------------------------------------------------------------------
template <int S, int U, int W, int D, int XOFF, int OOFF>
__global__ __launch_bounds__(256) void fctp_kernel(
    const float* __restrict__ x, const float* __restrict__ z,
    const _Float16* __restrict__ bw, float* __restrict__ out, int nNodes) {
  constexpr int T   = W / 16;
  constexpr int NPW = S * 128;           // 8 waves x S subtiles x 16 nodes

  __shared__ _Float16 zf[NPW * KPAD];    // z in A-fragment order, per node
  __shared__ _Float16 xs[U * D * NPW];   // x transposed: [q = u*D+m][nodeLocal]

  const int tid       = threadIdx.x;
  const int blockBase = blockIdx.x * NPW;

  // ---- stage z into A-fragment order (trip count is a multiple of 256) ----
  // A-fragment k order used here (contiguous, same as B packing): lanes 0-15 K={0..15}, lanes 16-31 K={16..31}
  for (int e = tid; e < NPW * KPAD; e += 256) {
    int nl  = e / KPAD;
    int idx = e % KPAD;
    int c = idx >> 5, r = idx & 31, h = r >> 4, j = r & 15;
    int k = c * 32 + h * 16 + j;
    int ng = blockBase + nl; if (ng >= nNodes) ng = nNodes - 1;   // clamp tail
    zf[e] = (k < NSP) ? (_Float16)z[(size_t)ng * NSP + k] : (_Float16)0.0f;
  }
  // ---- stage x transposed (U*D*NPW is a multiple of 256 for all blocks) ----
  for (int e = tid; e < U * D * NPW; e += 256) {
    int nl = e % NPW;
    int q  = e / NPW;
    int ng = blockBase + nl; if (ng >= nNodes) ng = nNodes - 1;
    xs[q * NPW + nl] = (_Float16)x[(size_t)ng * XDIM + XOFF + q];
  }
  __syncthreads();

  const int lane = tid & 31;
  const int wave = tid >> 5;
  const int col  = lane & 15;
  const int h    = lane >> 4;

  int nodeL[S];
#pragma unroll
  for (int s = 0; s < S; ++s) nodeL[s] = wave * (16 * S) + s * 16 + col;

  // Per-node z fragments (reused across all U iterations): 3 chunks x 16 f16
  v16h zfr[S][3];
#pragma unroll
  for (int s = 0; s < S; ++s)
#pragma unroll
    for (int c = 0; c < 3; ++c)
      zfr[s][c] = *(const v16h*)&zf[nodeL[s] * KPAD + c * 32 + h * 16];

  v8f acc[S][D][T];
#pragma unroll
  for (int s = 0; s < S; ++s)
#pragma unroll
    for (int m = 0; m < D; ++m)
#pragma unroll
      for (int t = 0; t < T; ++t) acc[s][m][t] = (v8f){};

#pragma unroll 2
  for (int u = 0; u < U; ++u) {
    _Float16 xv[S][D];
#pragma unroll
    for (int s = 0; s < S; ++s)
#pragma unroll
      for (int m = 0; m < D; ++m)
        xv[s][m] = xs[(u * D + m) * NPW + nodeL[s]];

#pragma unroll
    for (int c = 0; c < 3; ++c) {
      v16h A[S][D];
#pragma unroll
      for (int s = 0; s < S; ++s)
#pragma unroll
        for (int m = 0; m < D; ++m) A[s][m] = zfr[s][c] * splat16(xv[s][m]);

      const _Float16* bp = bw + (size_t)((u * 3 + c) * T) * 512 + lane * 16;
#pragma unroll
      for (int t = 0; t < T; ++t) {
        v16h B = *(const v16h*)(bp + t * 512);
#pragma unroll
        for (int s = 0; s < S; ++s)
#pragma unroll
          for (int m = 0; m < D; ++m)
            acc[s][m][t] = __builtin_amdgcn_wmma_f32_16x16x32_f16(
                false, A[s][m], false, B, (short)0, acc[s][m][t], false, false);
      }
    }
  }

  // ---- epilogue: alpha scale + f32 store ----
  // C/D layout: VGPR r, lanes 0-15 -> M=r, lanes 16-31 -> M=8+r; N = lane%16.
  const float alpha = rsqrtf((float)(U * NSP));
#pragma unroll
  for (int s = 0; s < S; ++s)
#pragma unroll
    for (int m = 0; m < D; ++m)
#pragma unroll
      for (int t = 0; t < T; ++t)
#pragma unroll
        for (int r = 0; r < 8; ++r) {
          int node = blockBase + wave * (16 * S) + s * 16 + r + 8 * h;
          if (node < nNodes)
            out[(size_t)node * XDIM + OOFF + (t * 16 + col) * D + m] =
                acc[s][m][t][r] * alpha;
        }
}

// ---------------------------------------------------------------------------
extern "C" void kernel_launch(void* const* d_in, const int* in_sizes, int n_in,
                              void* d_out, int out_size, void* d_ws, size_t ws_size,
                              hipStream_t stream) {
  const float* x  = (const float*)d_in[0];   // [N, 480]
  const float* z  = (const float*)d_in[1];   // [N, 89]
  const float* w0 = (const float*)d_in[2];   // [128, 89, 128]
  const float* w1 = (const float*)d_in[3];   // [64, 89, 64]
  const float* w2 = (const float*)d_in[4];   // [32, 89, 32]
  float*    out   = (float*)d_out;           // [N, 480]
  _Float16* bw    = (_Float16*)d_ws;         // ~4MB packed f16 weights

  const int nNodes = in_sizes[0] / XDIM;

  const int packElems = FRAGS_TOTAL * 512;
  pack_w_kernel<<<(packElems + 255) / 256, 256, 0, stream>>>(w0, w1, w2, bw, packElems);

  // l=0: dominant block -> 2 subtiles/wave, 256 nodes/WG (halves L2 B-traffic,
  // 2 WMMAs per B-fragment load). l=1/l=2: 1 subtile/wave (register pressure).
  const int grid0 = (nNodes + 255) / 256;
  const int grid1 = (nNodes + 127) / 128;
  fctp_kernel<2, 128, 128, 1,   0,   0><<<grid0, 256, 0, stream>>>(x, z, bw,                        out, nNodes);
  fctp_kernel<1,  64,  64, 3, 128, 128><<<grid1, 256, 0, stream>>>(x, z, bw + (size_t)FRAGS0 * 512, out, nNodes);
  fctp_kernel<1,  32,  32, 5, 320, 320><<<grid1, 256, 0, stream>>>(x, z, bw + (size_t)(FRAGS0 + FRAGS1) * 512, out, nNodes);
}
